// MultiHeadAttention_50912542327536
// MI455X (gfx1250) — matrix operand order repaired and checked by reading, not compiled, not run on hardware
//
#include <hip/hip_runtime.h>
#include <hip/hip_bf16.h>

typedef __bf16 bf16_t;
typedef __attribute__((ext_vector_type(16))) __bf16 v16bf;
typedef __attribute__((ext_vector_type(2)))  __bf16 v2bf;
typedef __attribute__((ext_vector_type(8)))  float  v8f;
typedef __attribute__((ext_vector_type(4)))  unsigned int uint4v;

#define D_MODEL 2048
#define N_HEADS 16
#define HEAD_DIM 128
#define SEQ 2048
#define BATCH 2
#define NTOK (BATCH * SEQ) /* 4096 */
#define ATTN_SCALE 0.08838834764831845f /* 1/sqrt(128), folded into q in rope_ln */

__device__ __forceinline__ bf16_t f2bf(float f) {
  unsigned u = __builtin_bit_cast(unsigned, f);
  u += 0x7FFFu + ((u >> 16) & 1u); // round-to-nearest-even
  unsigned short h = (unsigned short)(u >> 16);
  return __builtin_bit_cast(bf16_t, h);
}
__device__ __forceinline__ float bf2f(bf16_t b) {
  unsigned short h = __builtin_bit_cast(unsigned short, b);
  unsigned u = ((unsigned)h) << 16;
  return __builtin_bit_cast(float, u);
}

__device__ __forceinline__ v8f wmma_bf16(v16bf a, v16bf b, v8f c) {
  return __builtin_amdgcn_wmma_f32_16x16x32_bf16(false, a, false, b, (short)0, c,
                                                 false, false);
}

// DPP16 row rotation (rows = 16 lanes on wave32). ROW_ROR:n = 0x120 + n.
template <int CTRL>
__device__ __forceinline__ float dpp_mv(float x) {
  int xi = __builtin_bit_cast(int, x);
  int r = __builtin_amdgcn_update_dpp(xi, xi, CTRL, 0xf, 0xf, true);
  return __builtin_bit_cast(float, r);
}
// All-reduce across the 16-lane DPP row via rotations 1,2,4,8 (VALU only, no LDS).
__device__ __forceinline__ float rowmax16(float x) {
  x = fmaxf(x, dpp_mv<0x121>(x));
  x = fmaxf(x, dpp_mv<0x122>(x));
  x = fmaxf(x, dpp_mv<0x124>(x));
  x = fmaxf(x, dpp_mv<0x128>(x));
  return x;
}
__device__ __forceinline__ float rowsum16(float x) {
  x += dpp_mv<0x121>(x);
  x += dpp_mv<0x122>(x);
  x += dpp_mv<0x124>(x);
  x += dpp_mv<0x128>(x);
  return x;
}

// A fragment: logical A is 16x32 (MxK), stored row-major [rows, lda], K contiguous.
// Layout used here: lanes 0-15 = rows M, halfword h -> K=16g+h (contiguous per lane).
__device__ __forceinline__ v16bf load_A(const bf16_t* A, int lda, int m0, int k0) {
  int l = threadIdx.x & 31;
  int g = l >> 4;
  const bf16_t* p = A + (size_t)(m0 + (l & 15)) * lda + k0;
  v16bf a;
  ((uint4v*)&a)[0] = *(const uint4v*)(p + 16 * g);
  ((uint4v*)&a)[1] = *(const uint4v*)(p + 16 * g + 8);
  return a;
}
// B fragment: logical B is 32x16 (KxN) with B[k][n] = Brows[n][k] (row-major [N, ldb]).
// Layout used here: lane n = l&15, halfword h -> K = k0 + 16g + h (contiguous per lane).
__device__ __forceinline__ v16bf load_Bt(const bf16_t* B, int ldb, int n0, int k0) {
  int l = threadIdx.x & 31;
  int g = l >> 4;
  const bf16_t* p = B + (size_t)(n0 + (l & 15)) * ldb + k0 + 16 * g;
  v16bf b;
  ((uint4v*)&b)[0] = ((const uint4v*)p)[0];
  ((uint4v*)&b)[1] = ((const uint4v*)p)[1];
  return b;
}

// Stage a 128x64 bf16 tile (rows of stride K) into LDS via async-to-LDS DMA.
// 256 threads x 4 transfers of b128 = 16KB. Tracked on ASYNCcnt.
__device__ __forceinline__ void stage_B64_async(const bf16_t* Bsrc, bf16_t* ldsDst,
                                                int K) {
  int t = threadIdx.x;
#pragma unroll
  for (int j = 0; j < 4; ++j) {
    int idx = t + j * 256;
    int row = idx >> 3, seg = idx & 7;
    const bf16_t* gp = Bsrc + (size_t)row * K + seg * 8;
    unsigned lds_off = (unsigned)(size_t)(ldsDst + row * 64 + seg * 8);
    asm volatile("global_load_async_to_lds_b128 %0, %1, off"
                 :
                 : "v"(lds_off), "v"(gp)
                 : "memory");
  }
}
__device__ __forceinline__ void wait_async0() {
  asm volatile("s_wait_asynccnt 0" ::: "memory");
}

__global__ void cvt_f32_bf16(const float* __restrict__ src, bf16_t* __restrict__ dst,
                             int n) {
  int i = blockIdx.x * 256 + threadIdx.x;
  if (i < n) dst[i] = f2bf(src[i]);
}

// C[M,N] = A[M,K] @ B[N,K]^T.
// Block = 128(M) x 128(N), 8 waves in 4x2; wave computes 32x64 (8 accumulators).
// B tile (128x64) double-buffered in LDS via global_load_async_to_lds_b128.
template <bool OUT_BF16>
__global__ __launch_bounds__(256) void gemm_abt(const bf16_t* __restrict__ A,
                                                const bf16_t* __restrict__ B,
                                                void* __restrict__ Cp, int M, int N,
                                                int K) {
  __shared__ bf16_t ldsB[2][128 * 64];
  int wave = threadIdx.x >> 5;
  int l = threadIdx.x & 31;
  int g = l >> 4, n = l & 15;
  int wm = wave >> 1, wn = wave & 1;
  int m0 = blockIdx.y * 128 + wm * 32;
  int nblk = blockIdx.x * 128;
  const v8f z = {0.f, 0.f, 0.f, 0.f, 0.f, 0.f, 0.f, 0.f};
  v8f acc[2][4];
#pragma unroll
  for (int s = 0; s < 2; ++s)
#pragma unroll
    for (int j = 0; j < 4; ++j) acc[s][j] = z;

  const bf16_t* Bblk = B + (size_t)nblk * K;
  stage_B64_async(Bblk, &ldsB[0][0], K);
  int nsteps = K >> 6;
  for (int i = 0; i < nsteps; ++i) {
    wait_async0();   // my stage for buffer i&1 has landed
    __syncthreads(); // everyone's stage landed; prev readers done
    if (i + 1 < nsteps) {
      stage_B64_async(Bblk + 64 * (i + 1), &ldsB[(i + 1) & 1][0], K);
      // prefetch next A rows for this wave (speculative, no counters)
      __builtin_prefetch(A + (size_t)(m0 + l) * K + 64 * (i + 1), 0, 3);
    }
    const bf16_t* bb = &ldsB[i & 1][0];
#pragma unroll
    for (int kk = 0; kk < 64; kk += 32) {
      v16bf a0 = load_A(A, K, m0, 64 * i + kk);
      v16bf a1 = load_A(A, K, m0 + 16, 64 * i + kk);
#pragma unroll
      for (int j = 0; j < 4; ++j) {
        v16bf bfrag = load_Bt(bb, 64, wn * 64 + 16 * j, kk);
        acc[0][j] = wmma_bf16(a0, bfrag, acc[0][j]);
        acc[1][j] = wmma_bf16(a1, bfrag, acc[1][j]);
      }
    }
  }
#pragma unroll
  for (int s = 0; s < 2; ++s) {
#pragma unroll
    for (int j = 0; j < 4; ++j) {
#pragma unroll
      for (int r = 0; r < 8; ++r) {
        size_t idx =
            (size_t)(m0 + 16 * s + r + 8 * g) * N + (nblk + wn * 64 + 16 * j + n);
        if (OUT_BF16)
          ((bf16_t*)Cp)[idx] = f2bf(acc[s][j][r]);
        else
          ((float*)Cp)[idx] = acc[s][j][r];
      }
    }
  }
}

// Fused RoPE + per-head layernorm on q and k, in place (bf16 rows of 128).
// One wave per (token, head): lane handles freq pairs j=l and j=l+32.
// The attention scale 1/sqrt(128) is folded into q here.
__global__ __launch_bounds__(128) void rope_ln(bf16_t* __restrict__ qb,
                                               bf16_t* __restrict__ kb,
                                               const float* __restrict__ qw,
                                               const float* __restrict__ kw) {
  int wv = (blockIdx.x * 128 + threadIdx.x) >> 5; // 0..65535
  int l = threadIdx.x & 31;
  int h = wv & (N_HEADS - 1);
  int tok = wv >> 4;         // 0..4095
  int pos = tok & (SEQ - 1); // sequence position
  size_t base = (size_t)tok * D_MODEL + (size_t)h * HEAD_DIM;
  const float LOG_THETA = 9.210340371976184f; // ln(10000)
  float inv0 = __expf(-(float)(2 * l) * (LOG_THETA / 128.0f));
  float inv1 = __expf(-(float)(2 * l + 64) * (LOG_THETA / 128.0f));
  float a0 = (float)pos * inv0, a1 = (float)pos * inv1;
  float c0 = __cosf(a0), s0 = __sinf(a0);
  float c1 = __cosf(a1), s1 = __sinf(a1);
  for (int which = 0; which < 2; ++which) {
    bf16_t* buf = which ? kb : qb;
    const float* w = which ? kw : qw;
    v2bf p0 = *(const v2bf*)(buf + base + 2 * l);
    v2bf p1 = *(const v2bf*)(buf + base + 64 + 2 * l);
    float e0 = bf2f(p0[0]), o0 = bf2f(p0[1]);
    float e1 = bf2f(p1[0]), o1 = bf2f(p1[1]);
    float r0 = e0 * c0 - o0 * s0, r1 = e0 * s0 + o0 * c0;
    float r2 = e1 * c1 - o1 * s1, r3 = e1 * s1 + o1 * c1;
    float sum = r0 + r1 + r2 + r3;
#pragma unroll
    for (int off = 1; off < 32; off <<= 1) sum += __shfl_xor(sum, off);
    float mu = sum * (1.0f / 128.0f);
    float d0 = r0 - mu, d1 = r1 - mu, d2 = r2 - mu, d3 = r3 - mu;
    float vs = d0 * d0 + d1 * d1 + d2 * d2 + d3 * d3;
#pragma unroll
    for (int off = 1; off < 32; off <<= 1) vs += __shfl_xor(vs, off);
    float rs = rsqrtf(vs * (1.0f / 128.0f) + 1e-5f);
    float rsw = rs * (which ? 1.0f : ATTN_SCALE); // fold attn scale into q
    r0 = d0 * rsw * w[2 * l];
    r1 = d1 * rsw * w[2 * l + 1];
    r2 = d2 * rsw * w[64 + 2 * l];
    r3 = d3 * rsw * w[65 + 2 * l];
    v2bf q0v; q0v[0] = f2bf(r0); q0v[1] = f2bf(r1);
    v2bf q1v; q1v[0] = f2bf(r2); q1v[1] = f2bf(r3);
    *(v2bf*)(buf + base + 2 * l) = q0v;
    *(v2bf*)(buf + base + 64 + 2 * l) = q1v;
  }
}

// vT[b][h][d][s] = vb[b][s][h*128+d]  (so P@V B-fragments are row-contiguous)
__global__ void transpose_v(const bf16_t* __restrict__ vb, bf16_t* __restrict__ vT) {
  size_t i = (size_t)blockIdx.x * 256 + threadIdx.x;
  int s = (int)(i & (SEQ - 1));
  size_t r = i >> 11;
  int d = (int)(r & (HEAD_DIM - 1));
  size_t bh = r >> 7;
  int h = (int)(bh & (N_HEADS - 1));
  int b = (int)(bh >> 4);
  vT[i] = vb[((size_t)b * SEQ + s) * D_MODEL + (size_t)h * HEAD_DIM + d];
}

// Flash attention with sliding-window causal mask. Wave = one 16-row q tile.
// Masking is only materialized for the ~2 boundary chunks (wave-uniform branch).
// Row max/sum reductions use DPP16 row rotations (no LDS traffic).
__global__ __launch_bounds__(128) void attn_fa(const bf16_t* __restrict__ qb,
                                               const bf16_t* __restrict__ kb,
                                               const bf16_t* __restrict__ vT,
                                               bf16_t* __restrict__ attnb,
                                               const int* __restrict__ wsz) {
  __shared__ bf16_t ldsP[4][16 * 32];
  const int window = *wsz;
  int wave = threadIdx.x >> 5;
  int l = threadIdx.x & 31;
  int g = l >> 4, n = l & 15;
  int bh = blockIdx.y;
  int b = bh >> 4, h = bh & 15;
  int q0 = (blockIdx.x * 4 + wave) * 16;
  const bf16_t* qbase = qb + ((size_t)b * SEQ) * D_MODEL + (size_t)h * HEAD_DIM;
  const bf16_t* kbase = kb + ((size_t)b * SEQ) * D_MODEL + (size_t)h * HEAD_DIM;
  const bf16_t* vtb = vT + ((size_t)bh * HEAD_DIM) * SEQ;
  bf16_t* myP = &ldsP[wave][0];

  v16bf qf[4];
#pragma unroll
  for (int c = 0; c < 4; ++c) qf[c] = load_A(qbase, D_MODEL, q0, 32 * c);

  const v8f z = {0.f, 0.f, 0.f, 0.f, 0.f, 0.f, 0.f, 0.f};
  v8f o[8];
#pragma unroll
  for (int t = 0; t < 8; ++t) o[t] = z;
  float mrow[8], lrow[8];
#pragma unroll
  for (int r = 0; r < 8; ++r) { mrow[r] = -1e30f; lrow[r] = 0.f; }

  int kstart = q0 - window;
  if (kstart < 0) kstart = 0;
  kstart &= ~31;
  for (int kc = kstart; kc < q0 + 16; kc += 32) {
    int kn = kc + 32;
    if (kn < q0 + 16) { // prefetch next chunk of K rows and V columns
      __builtin_prefetch(kbase + (size_t)(kn + n) * D_MODEL + 32 * g, 0, 3);
#pragma unroll
      for (int pp = 0; pp < 4; ++pp)
        __builtin_prefetch(vtb + (size_t)(pp * 32 + l) * SEQ + kn, 0, 3);
    }
    v8f s0 = z, s1 = z;
#pragma unroll
    for (int c = 0; c < 4; ++c)
      s0 = wmma_bf16(qf[c], load_Bt(kbase, D_MODEL, kc, 32 * c), s0);
#pragma unroll
    for (int c = 0; c < 4; ++c)
      s1 = wmma_bf16(qf[c], load_Bt(kbase, D_MODEL, kc + 16, 32 * c), s1);
    // wave-uniform: chunk needs masking only at window-entry or diagonal
    bool need_mask = (kc + 31 > q0) || (kc < q0 + 15 - window);
    float scl[8];
#pragma unroll
    for (int r = 0; r < 8; ++r) {
      float a0 = s0[r];
      float a1 = s1[r];
      if (need_mask) {
        int m = q0 + r + 8 * g;
        int k0i = kc + n, k1i = kc + 16 + n;
        if (k0i > m || k0i < m - window) a0 = -1e30f;
        if (k1i > m || k1i < m - window) a1 = -1e30f;
      }
      float mx = rowmax16(fmaxf(a0, a1));
      float nm = fmaxf(mrow[r], mx);
      float sc = __expf(mrow[r] - nm);
      float p0 = __expf(a0 - nm);
      float p1 = __expf(a1 - nm);
      float ps = rowsum16(p0 + p1);
      lrow[r] = lrow[r] * sc + ps;
      mrow[r] = nm;
      scl[r] = sc;
      myP[(r + 8 * g) * 32 + n] = f2bf(p0);
      myP[(r + 8 * g) * 32 + 16 + n] = f2bf(p1);
    }
#pragma unroll
    for (int t = 0; t < 8; ++t)
#pragma unroll
      for (int r = 0; r < 8; ++r) o[t][r] *= scl[r];
    asm volatile("s_wait_dscnt 0" ::: "memory");
    v16bf pa = load_A(myP, 32, 0, 0);
#pragma unroll
    for (int t = 0; t < 8; ++t)
      o[t] = wmma_bf16(pa, load_Bt(vtb, SEQ, t * 16, kc), o[t]);
  }
  float inv[8];
#pragma unroll
  for (int r = 0; r < 8; ++r) inv[r] = __builtin_amdgcn_rcpf(lrow[r]);
  bf16_t* ob = attnb + ((size_t)b * SEQ + q0) * D_MODEL + (size_t)h * HEAD_DIM;
#pragma unroll
  for (int t = 0; t < 8; ++t)
#pragma unroll
    for (int r = 0; r < 8; ++r)
      ob[(size_t)(r + 8 * g) * D_MODEL + t * 16 + n] = f2bf(o[t][r] * inv[r]);
}

extern "C" void kernel_launch(void* const* d_in, const int* in_sizes, int n_in,
                              void* d_out, int out_size, void* d_ws, size_t ws_size,
                              hipStream_t stream) {
  const float* x  = (const float*)d_in[0];
  const float* wq = (const float*)d_in[1];
  const float* wk = (const float*)d_in[2];
  const float* wv = (const float*)d_in[3];
  const float* wo = (const float*)d_in[4];
  const float* qw = (const float*)d_in[5];
  const float* kw = (const float*)d_in[6];
  const int* wsz  = (const int*)d_in[7];

  const size_t NX = (size_t)NTOK * D_MODEL;    // 8,388,608
  const size_t NW = (size_t)D_MODEL * D_MODEL; // 4,194,304
  bf16_t* xb    = (bf16_t*)d_ws;
  bf16_t* wqb   = xb + NX;
  bf16_t* wkb   = wqb + NW;
  bf16_t* wvb   = wkb + NW;
  bf16_t* wob   = wvb + NW;
  bf16_t* qb    = wob + NW;
  bf16_t* kb    = qb + NX;
  bf16_t* vb    = kb + NX;
  bf16_t* vT    = vb + NX;
  bf16_t* attnb = vT + NX;

  cvt_f32_bf16<<<(int)((NX + 255) / 256), 256, 0, stream>>>(x, xb, (int)NX);
  cvt_f32_bf16<<<(int)((NW + 255) / 256), 256, 0, stream>>>(wq, wqb, (int)NW);
  cvt_f32_bf16<<<(int)((NW + 255) / 256), 256, 0, stream>>>(wk, wkb, (int)NW);
  cvt_f32_bf16<<<(int)((NW + 255) / 256), 256, 0, stream>>>(wv, wvb, (int)NW);
  cvt_f32_bf16<<<(int)((NW + 255) / 256), 256, 0, stream>>>(wo, wob, (int)NW);

  dim3 gg(D_MODEL / 128, NTOK / 128); // (16, 32)
  gemm_abt<true><<<gg, 256, 0, stream>>>(xb, wqb, qb, NTOK, D_MODEL, D_MODEL);
  gemm_abt<true><<<gg, 256, 0, stream>>>(xb, wkb, kb, NTOK, D_MODEL, D_MODEL);
  gemm_abt<true><<<gg, 256, 0, stream>>>(xb, wvb, vb, NTOK, D_MODEL, D_MODEL);

  rope_ln<<<(NTOK * N_HEADS) / 4, 128, 0, stream>>>(qb, kb, qw, kw);
  transpose_v<<<(int)(NX / 256), 256, 0, stream>>>(vb, vT);

  attn_fa<<<dim3(SEQ / 64, BATCH * N_HEADS), 128, 0, stream>>>(qb, kb, vT, attnb,
                                                               wsz);

  gemm_abt<false><<<gg, 256, 0, stream>>>(attnb, wob, d_out, NTOK, D_MODEL,
                                          D_MODEL);
}
